// MultiHeadSelfAttention_14053132992952
// MI455X (gfx1250) — matrix operand order repaired and checked by reading, not compiled, not run on hardware
//
#include <hip/hip_runtime.h>

typedef unsigned short u16;
typedef unsigned int   u32;
typedef unsigned long long u64;
typedef __attribute__((ext_vector_type(16))) __bf16 v16bf;
typedef __attribute__((ext_vector_type(8)))  float  v8f;

#define T_SEQ  2048
#define DMODEL 1024
#define NHEAD  16
#define DHEAD  64
#define BATCH  4
#define MROWS  (BATCH * T_SEQ)

__device__ __forceinline__ u16 f32_to_bf16(float f) {
  u32 u = __float_as_uint(f);
  u32 r = u + 0x7FFFu + ((u >> 16) & 1u);   // round-to-nearest-even
  return (u16)(r >> 16);
}

__device__ __forceinline__ float redmax16(float v) {
  v = fmaxf(v, __shfl_xor(v, 1, 32));
  v = fmaxf(v, __shfl_xor(v, 2, 32));
  v = fmaxf(v, __shfl_xor(v, 4, 32));
  v = fmaxf(v, __shfl_xor(v, 8, 32));
  return v;
}
__device__ __forceinline__ float redsum16(float v) {
  v += __shfl_xor(v, 1, 32);
  v += __shfl_xor(v, 2, 32);
  v += __shfl_xor(v, 4, 32);
  v += __shfl_xor(v, 8, 32);
  return v;
}

// CDNA5 async global->LDS DMA (ASYNCcnt-tracked), 16B per lane.
__device__ __forceinline__ void async_b128(u32 lds_addr, const u16* g) {
  asm volatile("global_load_async_to_lds_b128 %0, %1, off"
               :: "v"(lds_addr), "v"((u64)(size_t)g) : "memory");
}
__device__ __forceinline__ void wait_async0() {
  asm volatile("s_wait_asynccnt 0" ::: "memory");
}
__device__ __forceinline__ u32 lds_off(const void* p) {
  return (u32)(size_t)p;   // low 32 bits of generic LDS pointer = LDS byte address
}

// ---------------- f32 -> bf16 flat conversion (x4) ----------------
__global__ void cvt_f32_bf16(const float* __restrict__ in, u16* __restrict__ out, int n4) {
  int i = blockIdx.x * blockDim.x + threadIdx.x;
  if (i < n4) {
    float4 v = ((const float4*)in)[i];
    ushort4 o;
    o.x = f32_to_bf16(v.x); o.y = f32_to_bf16(v.y);
    o.z = f32_to_bf16(v.z); o.w = f32_to_bf16(v.w);
    ((ushort4*)out)[i] = o;
  }
}

// ---------------- f32 W[k][n] -> bf16 Wt[n][k] (LDS-tiled transpose) ----------------
__global__ __launch_bounds__(256) void cvt_transpose_bf16(const float* __restrict__ W,
                                                          u16* __restrict__ Wt) {
  __shared__ float tile[32][33];
  const int tx  = threadIdx.x & 31;
  const int ty4 = threadIdx.x >> 5;      // 0..7
  const int k0 = blockIdx.y * 32, n0 = blockIdx.x * 32;
#pragma unroll
  for (int r = 0; r < 4; ++r) {
    int ky = ty4 * 4 + r;
    tile[ky][tx] = W[(size_t)(k0 + ky) * DMODEL + n0 + tx];
  }
  __syncthreads();
#pragma unroll
  for (int r = 0; r < 4; ++r) {
    int ny = ty4 * 4 + r;
    Wt[(size_t)(n0 + ny) * DMODEL + k0 + tx] = f32_to_bf16(tile[tx][ny]);
  }
}

// ---------------- bf16 WMMA GEMM: C = A[M,K] * Wt[N,K]^T + bias ----------------
// Block tile 128x128, 8 waves -> wave tile 32(M) x 64(N) = 2x4 WMMA accumulators.
// Double-buffered LDS staging via global_load_async_to_lds_b128.
// MODE 0: bf16 out as Q/K layout [B,H,T,Dh]
// MODE 2: bf16 out as V-transposed [B,H,Dh,T]
// MODE 3: f32 out row-major [M,N]
template <int MODE>
__global__ __launch_bounds__(256) void gemm_bf16(
    const u16* __restrict__ A, const u16* __restrict__ Wt,
    const float* __restrict__ bias, void* __restrict__ Out,
    int M, int N, int K)
{
  __shared__ __align__(16) u16 As[2][128 * 32];   // [row][k]
  __shared__ __align__(16) u16 Bs[2][128 * 32];   // [n][k]

  const int tid   = threadIdx.x;
  const int lane  = tid & 31;
  const int wave  = tid >> 5;
  const int waveM = wave >> 1;       // 0..3 -> 32 rows each
  const int waveN = wave & 1;        // 0..1 -> 64 cols each
  const int nLane = lane & 15;
  const int hi8   = (lane >> 4) * 8;
  const int kst   = (lane >> 4) * 16;
  const int rowBase = blockIdx.y * 128;
  const int colBase = blockIdx.x * 128;

  const int srow = tid >> 1;          // 0..127
  const int skof = (tid & 1) * 16;    // 0 or 16

  auto stage = [&](int buf, int kk) {
    const u16* ga = A  + (size_t)(rowBase + srow) * K + kk + skof;
    const u16* gb = Wt + (size_t)(colBase + srow) * K + kk + skof;
    async_b128(lds_off(&As[buf][srow * 32 + skof]),     ga);
    async_b128(lds_off(&As[buf][srow * 32 + skof + 8]), ga + 8);
    async_b128(lds_off(&Bs[buf][srow * 32 + skof]),     gb);
    async_b128(lds_off(&Bs[buf][srow * 32 + skof + 8]), gb + 8);
  };

  v8f acc[2][4] = {};
  int buf = 0;
  stage(0, 0);

  for (int kk = 0; kk < K; kk += 32) {
    wait_async0();
    __syncthreads();
    if (kk + 32 < K) stage(buf ^ 1, kk + 32);

    v16bf afrag[2], bfrag[4];
#pragma unroll
    for (int tm = 0; tm < 2; ++tm) {
      const u16* ab = &As[buf][(waveM * 32 + tm * 16 + nLane) * 32 + hi8];
      ((uint4*)&afrag[tm])[0] = *(const uint4*)(ab);        // K = hi8 .. hi8+7
      ((uint4*)&afrag[tm])[1] = *(const uint4*)(ab + 16);   // K = hi8+16 .. hi8+23
    }
#pragma unroll
    for (int tn = 0; tn < 4; ++tn) {
      const u16* bb = &Bs[buf][(waveN * 64 + tn * 16 + nLane) * 32 + hi8];
      ((uint4*)&bfrag[tn])[0] = *(const uint4*)(bb);        // K = hi8 .. hi8+7
      ((uint4*)&bfrag[tn])[1] = *(const uint4*)(bb + 16);   // K = hi8+16 .. hi8+23
    }
#pragma unroll
    for (int tm = 0; tm < 2; ++tm)
#pragma unroll
      for (int tn = 0; tn < 4; ++tn)
        acc[tm][tn] = __builtin_amdgcn_wmma_f32_16x16x32_bf16(
            false, afrag[tm], false, bfrag[tn], (short)0, acc[tm][tn], false, false);
    __syncthreads();
    buf ^= 1;
  }

  // Epilogue: bias + layout-mapped store
#pragma unroll
  for (int tm = 0; tm < 2; ++tm) {
#pragma unroll
    for (int tn = 0; tn < 4; ++tn) {
      const int tileM = rowBase + waveM * 32 + tm * 16;
      const int tileN = colBase + waveN * 64 + tn * 16;
#pragma unroll
      for (int i = 0; i < 8; ++i) {
        const int m = tileM + hi8 + i;
        const int n = tileN + nLane;
        float v = acc[tm][tn][i] + bias[n];
        if (MODE == 3) {
          ((float*)Out)[(size_t)m * N + n] = v;
        } else {
          const int b = m / T_SEQ, t = m % T_SEQ;
          const int h = n / DHEAD, d = n % DHEAD;
          size_t idx;
          if (MODE == 2) idx = ((size_t)(b * NHEAD + h) * DHEAD + d) * T_SEQ + t;
          else           idx = ((size_t)(b * NHEAD + h) * T_SEQ + t) * DHEAD + d;
          ((u16*)Out)[idx] = f32_to_bf16(v);
        }
      }
    }
  }
}

// ---------------- Flash attention ----------------
// Q,K: [B*H, T, Dh] bf16. Vt: [B*H, Dh, T] bf16. ctx out: [B, T, D] bf16.
// One 16-query tile per wave; 32-key blocks; K/V tiles staged to LDS once per
// block (shared by all 8 waves) via async DMA, double-buffered.
__global__ __launch_bounds__(256) void attn_fa(
    const u16* __restrict__ Q, const u16* __restrict__ K,
    const u16* __restrict__ Vt, u16* __restrict__ ctx)
{
  __shared__ __align__(16) u16 Ks[2][32 * DHEAD];   // [key][d]
  __shared__ __align__(16) u16 Vs[2][DHEAD * 32];   // [d][key]
  __shared__ __align__(16) u16 Plds[8 * 16 * 32];   // per-wave P staging

  const int tid  = threadIdx.x;
  const int lane = tid & 31;
  const int wave = tid >> 5;
  const int nLane = lane & 15;
  const int hi8   = (lane >> 4) * 8;
  const int kst   = (lane >> 4) * 16;
  const int bh = blockIdx.y;
  const int q0 = blockIdx.x * 128 + wave * 16;

  const u16* Qp = Q  + (size_t)bh * T_SEQ * DHEAD;
  const u16* Kp = K  + (size_t)bh * T_SEQ * DHEAD;
  const u16* Vp = Vt + (size_t)bh * DHEAD * T_SEQ;
  u16* pw = &Plds[wave * 16 * 32];

  auto stageKV = [&](int buf, int n0) {
    {  // K tile: 32 keys x 64 d, contiguous in d
      int row = tid >> 3;            // 0..31
      int c8  = (tid & 7) * 8;       // 0..56
      async_b128(lds_off(&Ks[buf][row * DHEAD + c8]),
                 Kp + (size_t)(n0 + row) * DHEAD + c8);
    }
    {  // V tile: 64 d x 32 keys, contiguous in key
      int drow = tid >> 2;           // 0..63
      int c8   = (tid & 3) * 8;      // 0..24
      async_b128(lds_off(&Vs[buf][drow * 32 + c8]),
                 Vp + (size_t)drow * T_SEQ + n0 + c8);
    }
  };

  // Q A-fragments for the two Dh/32 chunks, loaded once from global
  v16bf aq[2];
#pragma unroll
  for (int c = 0; c < 2; ++c) {
    const u16* qb = Qp + (size_t)(q0 + nLane) * DHEAD + c * 32 + hi8;
    ((uint4*)&aq[c])[0] = *(const uint4*)(qb);
    ((uint4*)&aq[c])[1] = *(const uint4*)(qb + 16);
  }

  v8f accO[4] = {};
  float m8[8], l8[8];
#pragma unroll
  for (int i = 0; i < 8; ++i) { m8[i] = -3.0e30f; l8[i] = 0.0f; }

  int buf = 0;
  stageKV(0, 0);

  for (int n0 = 0; n0 < T_SEQ; n0 += 32) {
    wait_async0();
    __syncthreads();
    if (n0 + 32 < T_SEQ) stageKV(buf ^ 1, n0 + 32);

    // S = (Q K^T) / sqrt(Dh): two 16x16 f32 tiles
    v8f S[2];
#pragma unroll
    for (int t = 0; t < 2; ++t) {
      v8f s = {};
#pragma unroll
      for (int c = 0; c < 2; ++c) {
        v16bf bk;
        const u16* kb = &Ks[buf][(t * 16 + nLane) * DHEAD + c * 32 + hi8];
        ((uint4*)&bk)[0] = *(const uint4*)(kb);
        ((uint4*)&bk)[1] = *(const uint4*)(kb + 16);
        s = __builtin_amdgcn_wmma_f32_16x16x32_bf16(false, aq[c], false, bk,
                                                    (short)0, s, false, false);
      }
#pragma unroll
      for (int i = 0; i < 8; ++i) s[i] *= 0.125f;   // 1/sqrt(64)
      S[t] = s;
    }

    // Online softmax (row stats replicated across 16-lane halves)
#pragma unroll
    for (int i = 0; i < 8; ++i) {
      float rmax = redmax16(fmaxf(S[0][i], S[1][i]));
      float mn   = fmaxf(m8[i], rmax);
      float corr = __expf(m8[i] - mn);
      m8[i] = mn;
      float p0 = __expf(S[0][i] - mn);
      float p1 = __expf(S[1][i] - mn);
      S[0][i] = p0; S[1][i] = p1;
      float rs = redsum16(p0 + p1);
      l8[i] = l8[i] * corr + rs;
#pragma unroll
      for (int dt = 0; dt < 4; ++dt) accO[dt][i] *= corr;
    }

    // C-layout -> A-layout for P via per-wave private LDS
#pragma unroll
    for (int t = 0; t < 2; ++t)
#pragma unroll
      for (int i = 0; i < 8; ++i)
        pw[(hi8 + i) * 32 + t * 16 + nLane] = f32_to_bf16(S[t][i]);

    v16bf pA;
    const u16* pb = &pw[nLane * 32 + hi8];
    ((uint4*)&pA)[0] = *(const uint4*)(pb);
    ((uint4*)&pA)[1] = *(const uint4*)(pb + 16);

    // O += P @ V
#pragma unroll
    for (int dt = 0; dt < 4; ++dt) {
      v16bf bv;
      const u16* vb = &Vs[buf][(dt * 16 + nLane) * 32 + hi8];
      ((uint4*)&bv)[0] = *(const uint4*)(vb);
      ((uint4*)&bv)[1] = *(const uint4*)(vb + 16);
      accO[dt] = __builtin_amdgcn_wmma_f32_16x16x32_bf16(false, pA, false, bv,
                                                         (short)0, accO[dt], false, false);
    }
    __syncthreads();
    buf ^= 1;
  }

  // Normalize and scatter to ctx [B, T, D]
  const int b = bh / NHEAD, h = bh % NHEAD;
#pragma unroll
  for (int dt = 0; dt < 4; ++dt)
#pragma unroll
    for (int i = 0; i < 8; ++i) {
      const int t = q0 + hi8 + i;
      const int d = dt * 16 + nLane;
      float v = accO[dt][i] / l8[i];
      ctx[((size_t)(b * T_SEQ + t)) * DMODEL + h * DHEAD + d] = f32_to_bf16(v);
    }
}

// ---------------- host-side launch ----------------
extern "C" void kernel_launch(void* const* d_in, const int* in_sizes, int n_in,
                              void* d_out, int out_size, void* d_ws, size_t ws_size,
                              hipStream_t stream) {
  (void)in_sizes; (void)n_in; (void)out_size; (void)ws_size;
  const float* x  = (const float*)d_in[0];
  const float* Wq = (const float*)d_in[1];
  const float* bq = (const float*)d_in[2];
  const float* Wk = (const float*)d_in[3];
  const float* bk = (const float*)d_in[4];
  const float* Wv = (const float*)d_in[5];
  const float* bv = (const float*)d_in[6];
  const float* Wo = (const float*)d_in[7];
  const float* bo = (const float*)d_in[8];

  char* ws = (char*)d_ws;
  const size_t szX = (size_t)MROWS * DMODEL * sizeof(u16);    // 16 MB
  const size_t szW = (size_t)DMODEL * DMODEL * sizeof(u16);   // 2 MB
  u16* xb  = (u16*)(ws);
  u16* Wqb = (u16*)(ws + szX);                 // transposed [n][k]
  u16* Wkb = (u16*)(ws + szX + 1 * szW);
  u16* Wvb = (u16*)(ws + szX + 2 * szW);
  u16* Wob = (u16*)(ws + szX + 3 * szW);
  u16* Qb  = (u16*)(ws + szX + 4 * szW);
  u16* Kb  = (u16*)(ws + 2 * szX + 4 * szW);
  u16* Vtb = (u16*)(ws + 3 * szX + 4 * szW);
  u16* ctx = (u16*)(ws + 4 * szX + 4 * szW);

  const int n4x = MROWS * DMODEL / 4;
  cvt_f32_bf16<<<(n4x + 255) / 256, 256, 0, stream>>>(x, xb, n4x);
  dim3 gt(DMODEL / 32, DMODEL / 32);
  cvt_transpose_bf16<<<gt, 256, 0, stream>>>(Wq, Wqb);
  cvt_transpose_bf16<<<gt, 256, 0, stream>>>(Wk, Wkb);
  cvt_transpose_bf16<<<gt, 256, 0, stream>>>(Wv, Wvb);
  cvt_transpose_bf16<<<gt, 256, 0, stream>>>(Wo, Wob);

  dim3 gg(DMODEL / 128, MROWS / 128);
  gemm_bf16<0><<<gg, 256, 0, stream>>>(xb, Wqb, bq, Qb,  MROWS, DMODEL, DMODEL);
  gemm_bf16<0><<<gg, 256, 0, stream>>>(xb, Wkb, bk, Kb,  MROWS, DMODEL, DMODEL);
  gemm_bf16<2><<<gg, 256, 0, stream>>>(xb, Wvb, bv, Vtb, MROWS, DMODEL, DMODEL);

  attn_fa<<<dim3(T_SEQ / 128, BATCH * NHEAD), 256, 0, stream>>>(Qb, Kb, Vtb, ctx);

  gemm_bf16<3><<<gg, 256, 0, stream>>>(ctx, Wob, bo, d_out, MROWS, DMODEL, DMODEL);
}
